// TransformerNet_10316511445143
// MI455X (gfx1250) — compile-verified, matrix operand order checked by reading, not run on hardware
//
#include <hip/hip_runtime.h>
#include <hip/hip_bf16.h>

// ---------------------------------------------------------------------------
// GPT forward (6 layers, B=4,T=1024,C=768,NH=12,HD=64,HID=3072,V=32000)
// for MI455X / gfx1250.
//   * all GEMMs: v_wmma_f32_16x16x32_bf16, 256x64 block tile, 8 waves,
//     wave tile 64x32 (acc[4][2]).
//   * K-stages double-buffered in LDS via GLOBAL_LOAD_ASYNC_TO_LDS_B128
//     (ASYNCcnt, 5 async ops/stage) -> no VGPR staging, loads overlap WMMA.
// Workspace: ~530 MB (activations, scores/probs, bf16 weight copies).
// ---------------------------------------------------------------------------

typedef unsigned short u16;
typedef __attribute__((ext_vector_type(16))) __bf16 v16bf;
typedef __attribute__((ext_vector_type(8)))  float  v8f;

#define Lc   6
#define Cc   768
#define HIDc 3072
#define NHc  12
#define Vc   32000
#define Bc   4
#define Tc   1024
#define HDc  64
#define Mc   (Bc * Tc)          // 4096 activation rows

#define A_ELEMS (256 * 40)      // one A buffer: 256 rows x (32+8 pad) halves
#define B_ELEMS (64 * 40)       // one B buffer:  64 rows x (32+8 pad) halves

__device__ __forceinline__ u16 f2bf(float f) {
    unsigned u = __float_as_uint(f);
    return (u16)((u + 0x7fffu + ((u >> 16) & 1u)) >> 16);   // RNE truncate
}
__device__ __forceinline__ float gelu_exact(float x) {
    return 0.5f * x * (1.0f + erff(x * 0.70710678118654752f));
}
__device__ __forceinline__ unsigned lds_addr32(const void* p) {
    // generic pointer to LDS: low 32 bits are the LDS offset (ISA 10.2)
    return (unsigned)(unsigned long long)p;
}
__device__ __forceinline__ void async_load_b128(unsigned lds, const void* gaddr) {
    asm volatile("global_load_async_to_lds_b128 %0, %1, off"
                 :: "v"(lds), "v"((unsigned long long)gaddr) : "memory");
}

union FragU { v16bf v; uint4 q[2]; };

// ---------------------------------------------------------------------------
// WMMA GEMM:  D = epi( alpha * A @ B^T + bias ) [+ resid]
//   A: [M,K] bf16 row-major, B: [N,K] bf16 row-major.
//   Batched over blockIdx.z with (outer,inner) strides split by nhdiv.
//   flags: 1=gelu, 2=+resid(f32), 4=+bias, 8=store f32 Cf, 16=store bf16 Cb.
// Block tile 256x64 (8 waves, each 64x32 = 4x2 wmma accums).
// M%256==0, N%64==0, K%32==0 guaranteed by caller -> no bounds checks,
// EXEC all-ones around WMMA as required.
// K-loop: double-buffered LDS; next stage loaded with async-to-LDS b128
// (4 instr for A, 1 for B), s_wait_asynccnt + barrier.
// All ld* are multiples of 8 elements and tiles start at multiples of 8, so
// every global/LDS address is 16B-aligned (LDS row pitch 80B, chunks at
// {0,16,32,48}B).
// ---------------------------------------------------------------------------
__global__ __launch_bounds__(256) void k_gemm(
    const u16* __restrict__ A, const u16* __restrict__ Bm,
    const float* __restrict__ bias, const float* __restrict__ resid,
    float* __restrict__ Cf, u16* __restrict__ Cb,
    int K, int lda, int ldb, int ldc,
    float alpha, int flags, int nhdiv,
    long long a_os, long long a_is, long long b_os, long long b_is,
    long long c_os, long long c_is)
{
    __shared__ u16 Al[2 * A_ELEMS];   // [buf][m][k] padded rows (80B pitch)
    __shared__ u16 Bl[2 * B_ELEMS];   // [buf][n][k] padded rows

    const int tid  = threadIdx.x;
    const int lane = tid & 31;
    const int wave = tid >> 5;
    const int wm   = (wave & 3) << 6;    // wave row offset (0,64,128,192)
    const int wn   = (wave >> 2) << 5;   // wave col offset (0,32)
    const int m0   = blockIdx.y << 8;
    const int n0   = blockIdx.x << 6;
    const int z    = blockIdx.z;

    const long long aoff = (long long)(z / nhdiv) * a_os + (long long)(z % nhdiv) * a_is;
    const long long boff = (long long)(z / nhdiv) * b_os + (long long)(z % nhdiv) * b_is;
    const long long coff = (long long)(z / nhdiv) * c_os + (long long)(z % nhdiv) * c_is;

    // issue one K-stage (32 wide) of async memory->LDS copies (b128 = 8 halves)
    auto issue = [&](int k0, int buf) {
#pragma unroll
        for (int it = 0; it < 4; ++it) {           // A: 256 rows x 4 chunks
            int ch = tid + (it << 8);
            int r = ch >> 2, c8 = (ch & 3) << 3;
            async_load_b128(lds_addr32(&Al[buf * A_ELEMS + r * 40 + c8]),
                            &A[aoff + (long long)(m0 + r) * lda + k0 + c8]);
        }
        {                                          // B: 64 rows x 4 chunks
            int r = tid >> 2, c8 = (tid & 3) << 3;
            async_load_b128(lds_addr32(&Bl[buf * B_ELEMS + r * 40 + c8]),
                            &Bm[boff + (long long)(n0 + r) * ldb + k0 + c8]);
        }
    };

    v8f acc[4][2];
#pragma unroll
    for (int i = 0; i < 4; ++i)
#pragma unroll
        for (int j = 0; j < 2; ++j)
#pragma unroll
            for (int e = 0; e < 8; ++e) acc[i][j][e] = 0.0f;

    const int nst = K >> 5;
    issue(0, 0);
    for (int s = 0; s < nst; ++s) {
        const int cur = s & 1;
        if (s + 1 < nst) {
            issue((s + 1) << 5, cur ^ 1);
            // stage s complete once <=5 (stage s+1's) async ops outstanding
            asm volatile("s_wait_asynccnt 0x5" ::: "memory");
        } else {
            asm volatile("s_wait_asynccnt 0x0" ::: "memory");
        }
        __syncthreads();

        // fragments (ISA 16-bit layout: lanes 0-15 hold K{0..7,16..23},
        // lanes 16-31 hold K{8..15,24..31}) -> two b128 LDS loads each
        const u16* Ab = &Al[cur * A_ELEMS];
        const u16* Bb = &Bl[cur * B_ELEMS];
        const int klo  = (lane >> 4) << 3;
        const int ln15 = lane & 15;
        FragU fa[4], fb[2];
#pragma unroll
        for (int i = 0; i < 4; ++i) {
            const u16* p = &Ab[(wm + (i << 4) + ln15) * 40 + klo];
            fa[i].q[0] = *reinterpret_cast<const uint4*>(p);
            fa[i].q[1] = *reinterpret_cast<const uint4*>(p + 16);
        }
#pragma unroll
        for (int j = 0; j < 2; ++j) {
            const u16* p = &Bb[(wn + (j << 4) + ln15) * 40 + klo];
            fb[j].q[0] = *reinterpret_cast<const uint4*>(p);
            fb[j].q[1] = *reinterpret_cast<const uint4*>(p + 16);
        }
#pragma unroll
        for (int i = 0; i < 4; ++i)
#pragma unroll
            for (int j = 0; j < 2; ++j)
                acc[i][j] = __builtin_amdgcn_wmma_f32_16x16x32_bf16(
                    false, fa[i].v, false, fb[j].v,
                    (short)0, acc[i][j], false, false);
        __syncthreads();   // reads done before buffer `cur` is overwritten
    }

    // epilogue (C/D layout: VGPR r -> row r (+8 for lanes>=16), N = lane&15)
    const int hi   = (lane >> 4) << 3;
    const int ln15 = lane & 15;
#pragma unroll
    for (int i = 0; i < 4; ++i)
#pragma unroll
        for (int j = 0; j < 2; ++j)
#pragma unroll
            for (int r = 0; r < 8; ++r) {
                int gm = m0 + wm + (i << 4) + hi + r;
                int gn = n0 + wn + (j << 4) + ln15;
                float v = acc[i][j][r] * alpha;
                if (flags & 4)  v += bias[gn];
                if (flags & 1)  v = gelu_exact(v);
                long long idx = coff + (long long)gm * ldc + gn;
                if (flags & 2)  v += resid[idx];
                if (flags & 8)  Cf[idx] = v;
                if (flags & 16) Cb[idx] = f2bf(v);
            }
}

// ---------------------------------------------------------------------------
// Embedding + sinusoidal positional encoding -> fp32 residual stream x
// ---------------------------------------------------------------------------
__global__ __launch_bounds__(256) void k_embed(
    const int* __restrict__ tok, const float* __restrict__ emb,
    float* __restrict__ x)
{
    const int row = blockIdx.x;        // b*T + t
    const int tid = threadIdx.x;
    const int t   = row & (Tc - 1);
    const int tk  = tok[row];
#pragma unroll
    for (int j = 0; j < 3; ++j) {
        int c  = tid + (j << 8);
        int jj = (c < Cc / 2) ? c : (c - Cc / 2);
        float div = expf((float)jj * (-9.210340371976184f / (float)(Cc / 2 - 1)));
        float ang = (float)t * div;
        float pe  = (c < Cc / 2) ? sinf(ang) : cosf(ang);
        x[(long long)row * Cc + c] = emb[(long long)tk * Cc + c] + pe;
    }
}

// ---------------------------------------------------------------------------
// LayerNorm (population var, eps=1e-6): fp32 [rows,768] -> bf16 out
// ---------------------------------------------------------------------------
__global__ __launch_bounds__(256) void k_layernorm(
    const float* __restrict__ x, const float* __restrict__ w,
    const float* __restrict__ b, u16* __restrict__ out)
{
    __shared__ float red[256];
    __shared__ float st[2];
    const int row = blockIdx.x, tid = threadIdx.x;
    const float* xr = x + (long long)row * Cc;
    float v[3], s = 0.f, ss = 0.f;
#pragma unroll
    for (int j = 0; j < 3; ++j) {
        v[j] = xr[tid + (j << 8)];
        s += v[j]; ss += v[j] * v[j];
    }
    red[tid] = s; __syncthreads();
    for (int o = 128; o; o >>= 1) { if (tid < o) red[tid] += red[tid + o]; __syncthreads(); }
    if (!tid) st[0] = red[0] * (1.0f / Cc);
    __syncthreads();
    red[tid] = ss; __syncthreads();
    for (int o = 128; o; o >>= 1) { if (tid < o) red[tid] += red[tid + o]; __syncthreads(); }
    if (!tid) st[1] = red[0] * (1.0f / Cc) - st[0] * st[0];
    __syncthreads();
    const float mean = st[0], rstd = rsqrtf(st[1] + 1e-6f);
#pragma unroll
    for (int j = 0; j < 3; ++j) {
        int c = tid + (j << 8);
        out[(long long)row * Cc + c] = f2bf((v[j] - mean) * rstd * w[c] + b[c]);
    }
}

// ---------------------------------------------------------------------------
// Row softmax over T=1024: fp32 scores -> bf16 probs
// ---------------------------------------------------------------------------
__global__ __launch_bounds__(256) void k_softmax(
    const float* __restrict__ S, u16* __restrict__ P)
{
    __shared__ float red[256];
    __shared__ float st;
    const int row = blockIdx.x, tid = threadIdx.x;
    const float* sr = S + (long long)row * Tc;
    float v[4], mx = -3.402823466e+38f;
#pragma unroll
    for (int j = 0; j < 4; ++j) { v[j] = sr[tid + (j << 8)]; mx = fmaxf(mx, v[j]); }
    red[tid] = mx; __syncthreads();
    for (int o = 128; o; o >>= 1) { if (tid < o) red[tid] = fmaxf(red[tid], red[tid + o]); __syncthreads(); }
    if (!tid) st = red[0];
    __syncthreads();
    const float m = st;
    float s = 0.f;
#pragma unroll
    for (int j = 0; j < 4; ++j) { v[j] = expf(v[j] - m); s += v[j]; }
    red[tid] = s; __syncthreads();
    for (int o = 128; o; o >>= 1) { if (tid < o) red[tid] += red[tid + o]; __syncthreads(); }
    if (!tid) st = red[0];
    __syncthreads();
    const float inv = 1.0f / st;
#pragma unroll
    for (int j = 0; j < 4; ++j)
        P[(long long)row * Tc + tid + (j << 8)] = f2bf(v[j] * inv);
}

// ---------------------------------------------------------------------------
// V [B*T, C] (heads interleaved) -> Vt [B*NH, HD, T]  (bf16, for A·B^T PV)
// ---------------------------------------------------------------------------
__global__ __launch_bounds__(256) void k_transpose_v(
    const u16* __restrict__ vin, u16* __restrict__ vt)
{
    const long long ntot = (long long)Bc * NHc * HDc * Tc;
    long long o = (long long)blockIdx.x * 256 + threadIdx.x;
    const long long stride = (long long)gridDim.x * 256;
    for (; o < ntot; o += stride) {
        int t = (int)(o & (Tc - 1));
        long long r = o >> 10;            // z*HD + d
        int d = (int)(r & (HDc - 1));
        int zz = (int)(r >> 6);           // b*NH + h
        int b = zz / NHc, h = zz % NHc;
        vt[o] = vin[(long long)(b * Tc + t) * Cc + h * HDc + d];
    }
}

// ---------------------------------------------------------------------------
// fp32 -> bf16 weight conversion
// ---------------------------------------------------------------------------
__global__ __launch_bounds__(256) void k_f32_to_bf16(
    const float* __restrict__ in, u16* __restrict__ out, long long n)
{
    long long i = (long long)blockIdx.x * 256 + threadIdx.x;
    const long long stride = (long long)gridDim.x * 256;
    for (; i < n; i += stride) out[i] = f2bf(in[i]);
}

// ---------------------------------------------------------------------------
extern "C" void kernel_launch(void* const* d_in, const int* in_sizes, int n_in,
                              void* d_out, int out_size, void* d_ws, size_t ws_size,
                              hipStream_t stream)
{
    (void)in_sizes; (void)n_in; (void)out_size; (void)ws_size;

    const int*   tokens = (const int*)  d_in[0];
    const float* emb    = (const float*)d_in[1];
    const float* ln1w   = (const float*)d_in[2];
    const float* ln1b   = (const float*)d_in[3];
    const float* ln2w   = (const float*)d_in[4];
    const float* ln2b   = (const float*)d_in[5];
    const float* wq     = (const float*)d_in[6];
    const float* wk     = (const float*)d_in[7];
    const float* wv     = (const float*)d_in[8];
    const float* wp     = (const float*)d_in[9];
    const float* w1     = (const float*)d_in[10];
    const float* b1     = (const float*)d_in[11];
    const float* w2     = (const float*)d_in[12];
    const float* b2     = (const float*)d_in[13];
    const float* lnfw   = (const float*)d_in[14];
    const float* lnfb   = (const float*)d_in[15];
    const float* d1w    = (const float*)d_in[16];
    const float* d1b    = (const float*)d_in[17];
    const float* d2w    = (const float*)d_in[18];
    const float* d2b    = (const float*)d_in[19];
    const float* d3w    = (const float*)d_in[20];
    const float* d3b    = (const float*)d_in[21];

    // ---- carve workspace (256B aligned) ----
    size_t off = 0;
    auto carve = [&](size_t bytes) -> char* {
        char* p = (char*)d_ws + off;
        off += (bytes + 255) & ~(size_t)255;
        return p;
    };
    float* x    = (float*)carve((size_t)Mc * Cc * 4);       // fp32 residual stream
    u16*   acta = (u16*)  carve((size_t)Mc * HIDc * 2);     // bf16 act ping
    u16*   actb = (u16*)  carve((size_t)Mc * HIDc * 2);     // bf16 act pong
    u16*   qbuf = (u16*)  carve((size_t)Mc * Cc * 2);
    u16*   kbuf = (u16*)  carve((size_t)Mc * Cc * 2);
    u16*   vbuf = (u16*)  carve((size_t)Mc * Cc * 2);
    u16*   vtb  = (u16*)  carve((size_t)Bc * NHc * HDc * Tc * 2);  // V^T per head
    u16*   ybuf = (u16*)  carve((size_t)Mc * Cc * 2);
    float* Sbuf = (float*)carve((size_t)Bc * NHc * Tc * Tc * 4);   // attn scores
    u16*   Pbuf = (u16*)  carve((size_t)Bc * NHc * Tc * Tc * 2);   // attn probs
    u16*   wqb  = (u16*)  carve((size_t)Lc * Cc * Cc * 2);
    u16*   wkb  = (u16*)  carve((size_t)Lc * Cc * Cc * 2);
    u16*   wvb  = (u16*)  carve((size_t)Lc * Cc * Cc * 2);
    u16*   wpb  = (u16*)  carve((size_t)Lc * Cc * Cc * 2);
    u16*   w1b  = (u16*)  carve((size_t)Lc * HIDc * Cc * 2);
    u16*   w2b  = (u16*)  carve((size_t)Lc * Cc * HIDc * 2);
    u16*   d1wb = (u16*)  carve((size_t)Cc * Cc * 2);
    u16*   d2wb = (u16*)  carve((size_t)Cc * Cc * 2);
    u16*   d3wb = (u16*)  carve((size_t)Vc * Cc * 2);

    auto conv = [&](const float* src, u16* dst, long long n) {
        k_f32_to_bf16<<<2048, 256, 0, stream>>>(src, dst, n);
    };
    conv(wq, wqb, (long long)Lc * Cc * Cc);
    conv(wk, wkb, (long long)Lc * Cc * Cc);
    conv(wv, wvb, (long long)Lc * Cc * Cc);
    conv(wp, wpb, (long long)Lc * Cc * Cc);
    conv(w1, w1b, (long long)Lc * HIDc * Cc);
    conv(w2, w2b, (long long)Lc * Cc * HIDc);
    conv(d1w, d1wb, (long long)Cc * Cc);
    conv(d2w, d2wb, (long long)Cc * Cc);
    conv(d3w, d3wb, (long long)Vc * Cc);

    k_embed<<<Mc, 256, 0, stream>>>(tokens, emb, x);

    auto gemm = [&](const u16* Ap, const u16* Bp, const float* bias,
                    const float* resid, float* Cf, u16* Cb,
                    int M, int N, int K, int lda, int ldb, int ldc,
                    float alpha, int flags, int Z, int nhdiv,
                    long long a_os, long long a_is,
                    long long b_os, long long b_is,
                    long long c_os, long long c_is) {
        dim3 g(N / 64, M / 256, Z);
        k_gemm<<<g, 256, 0, stream>>>(Ap, Bp, bias, resid, Cf, Cb,
                                      K, lda, ldb, ldc, alpha, flags, nhdiv,
                                      a_os, a_is, b_os, b_is, c_os, c_is);
    };

    const float scale = 0.125f;  // 1/sqrt(HD=64)
    for (int i = 0; i < Lc; ++i) {
        // h = ln1(x)
        k_layernorm<<<Mc, 256, 0, stream>>>(x, ln1w + i * Cc, ln1b + i * Cc, acta);
        // q,k,v = h @ W^T (bf16 out)
        gemm(acta, wqb + (size_t)i * Cc * Cc, nullptr, nullptr, nullptr, qbuf,
             Mc, Cc, Cc, Cc, Cc, Cc, 1.f, 16, 1, 1, 0, 0, 0, 0, 0, 0);
        gemm(acta, wkb + (size_t)i * Cc * Cc, nullptr, nullptr, nullptr, kbuf,
             Mc, Cc, Cc, Cc, Cc, Cc, 1.f, 16, 1, 1, 0, 0, 0, 0, 0, 0);
        gemm(acta, wvb + (size_t)i * Cc * Cc, nullptr, nullptr, nullptr, vbuf,
             Mc, Cc, Cc, Cc, Cc, Cc, 1.f, 16, 1, 1, 0, 0, 0, 0, 0, 0);
        // S[b,h] = scale * Q[b,h] @ K[b,h]^T  (batched over 48 heads)
        gemm(qbuf, kbuf, nullptr, nullptr, Sbuf, nullptr,
             Tc, Tc, HDc, Cc, Cc, Tc, scale, 8, Bc * NHc, NHc,
             (long long)Tc * Cc, HDc, (long long)Tc * Cc, HDc,
             (long long)NHc * Tc * Tc, (long long)Tc * Tc);
        // P = softmax(S) -> bf16
        k_softmax<<<Bc * NHc * Tc, 256, 0, stream>>>(Sbuf, Pbuf);
        // Vt = V^T per head; Y[b,h] = P[b,h] @ Vt[b,h]^T
        k_transpose_v<<<4096, 256, 0, stream>>>(vbuf, vtb);
        gemm(Pbuf, vtb, nullptr, nullptr, nullptr, ybuf,
             Tc, HDc, Tc, Tc, Tc, Cc, 1.f, 16, Bc * NHc, NHc,
             (long long)NHc * Tc * Tc, (long long)Tc * Tc,
             (long long)NHc * HDc * Tc, (long long)HDc * Tc,
             (long long)Tc * Cc, HDc);
        // x = x + Y @ Wp^T
        gemm(ybuf, wpb + (size_t)i * Cc * Cc, nullptr, x, x, nullptr,
             Mc, Cc, Cc, Cc, Cc, Cc, 1.f, 8 | 2, 1, 1, 0, 0, 0, 0, 0, 0);
        // MLP: h = gelu(ln2(x) @ W1^T + b1);  x = x + h @ W2^T + b2
        k_layernorm<<<Mc, 256, 0, stream>>>(x, ln2w + i * Cc, ln2b + i * Cc, acta);
        gemm(acta, w1b + (size_t)i * HIDc * Cc, b1 + i * HIDc, nullptr, nullptr, actb,
             Mc, HIDc, Cc, Cc, Cc, HIDc, 1.f, 16 | 4 | 1, 1, 1, 0, 0, 0, 0, 0, 0);
        gemm(actb, w2b + (size_t)i * Cc * HIDc, b2 + i * Cc, x, x, nullptr,
             Mc, Cc, HIDc, HIDc, HIDc, Cc, 1.f, 8 | 4 | 2, 1, 1, 0, 0, 0, 0, 0, 0);
    }

    // head: lnf -> gelu(d1) -> gelu(d2) -> logits (fp32 to d_out)
    k_layernorm<<<Mc, 256, 0, stream>>>(x, lnfw, lnfb, acta);
    gemm(acta, d1wb, d1b, nullptr, nullptr, actb,
         Mc, Cc, Cc, Cc, Cc, Cc, 1.f, 16 | 4 | 1, 1, 1, 0, 0, 0, 0, 0, 0);
    gemm(actb, d2wb, d2b, nullptr, nullptr, acta,
         Mc, Cc, Cc, Cc, Cc, Cc, 1.f, 16 | 4 | 1, 1, 1, 0, 0, 0, 0, 0, 0);
    gemm(acta, d3wb, d3b, nullptr, (float*)d_out, nullptr,
         Mc, Vc, Cc, Cc, Cc, Vc, 1.f, 8 | 4, 1, 1, 0, 0, 0, 0, 0, 0);
}
